// AttentionBlock_10161892622607
// MI455X (gfx1250) — matrix operand order repaired and checked by reading, not compiled, not run on hardware
//
#include <hip/hip_runtime.h>
#include <hip/hip_bf16.h>

typedef __attribute__((ext_vector_type(16))) __bf16 v16bf;
typedef __attribute__((ext_vector_type(8)))  float  v8f;
typedef __attribute__((ext_vector_type(4)))  unsigned int u32x4;
typedef __attribute__((ext_vector_type(8)))  int i32x8;
typedef __attribute__((ext_vector_type(4)))  int i32x4;

constexpr int Bn = 8;
constexpr int Cc = 512;
constexpr int Tt = 1024;    // H*W
constexpr int NHh = 8;
constexpr int Gg = 32;
constexpr int HDd = 64;     // head dim

#ifndef __has_builtin
#define __has_builtin(x) 0
#endif
#if defined(__AMDGCN__) && __has_builtin(__builtin_amdgcn_tensor_load_to_lds) && \
    __has_builtin(__builtin_amdgcn_s_wait_tensorcnt)
#define USE_TDM 1
#else
#define USE_TDM 0
#endif

// ---------------------------------------------------------------------------
// WMMA helper (CDNA5 gfx1250, wave32). D = A(16x32 bf16) * B(32x16 bf16) + C(f32)
// ---------------------------------------------------------------------------
__device__ __forceinline__ v8f wmma_bf16(v16bf a, v16bf b, v8f c) {
  return __builtin_amdgcn_wmma_f32_16x16x32_bf16(
      /*neg_a=*/false, a, /*neg_b=*/false, b,
      /*c_mod=*/(short)0, c, /*reuse_a=*/false, /*reuse_b=*/false);
}

union Frag16 { v16bf v; uint4 q[2]; };

// A fragment (16x32) from row-major LDS tile p[m*ld + k], ld % 8 == 0.
__device__ __forceinline__ v16bf load_frag_a(const __bf16* p, int ld, int lane) {
  int m  = lane & 15;
  int hf = lane >> 4;
  const __bf16* base = p + m * ld + hf * 8;
  Frag16 f;
  f.q[0] = *(const uint4*)(base);        // K: hf*8 .. hf*8+7
  f.q[1] = *(const uint4*)(base + 16);   // K: 16+hf*8 .. 16+hf*8+7
  return f.v;
}

// B fragment (32x16, K x N) from *transposed* LDS tile pt[n*ld + k], ld % 8 == 0.
__device__ __forceinline__ v16bf load_frag_bt(const __bf16* pt, int ld, int lane) {
  int n  = lane & 15;
  int hf = lane >> 4;
  const __bf16* base = pt + n * ld + hf * 8;
  Frag16 f;
  f.q[0] = *(const uint4*)(base);        // K: hf*8 .. hf*8+7
  f.q[1] = *(const uint4*)(base + 16);   // K: 16+hf*8 .. 16+hf*8+7
  return f.v;
}

// 8x8 bf16 block transpose: global row-major [k][n] -> LDS [n][k].
__device__ __forceinline__ void xpose8x8(const __bf16* __restrict__ g, int gld,
                                         __bf16* __restrict__ l, int lld) {
  union { uint4 q; unsigned short s[8]; } in[8], out[8];
#pragma unroll
  for (int k = 0; k < 8; ++k) in[k].q = *(const uint4*)(g + (size_t)k * gld);
#pragma unroll
  for (int n = 0; n < 8; ++n) {
#pragma unroll
    for (int k = 0; k < 8; ++k) out[n].s[k] = in[k].s[n];
    *(uint4*)(l + n * lld) = out[n].q;
  }
}

#if USE_TDM
// -----------------------------------------------------------------------------
// TDM 2D tile load: global row-major [tile1][tile0] (tile0 contiguous, rows at
// stride0 elements, 2-byte elements) -> LDS at lds_byte_addr with LDS padding
// (pad_interval/pad_amount codes per D# group1). Issue from ONE wave; pair with
// s_wait_tensorcnt + block barrier.
// -----------------------------------------------------------------------------
__device__ __forceinline__ void tdm_load_2d(const void* g, unsigned lds_addr,
                                            unsigned tile0, unsigned tile1,
                                            unsigned stride0,
                                            unsigned padi, unsigned pada) {
  unsigned long long ga = (unsigned long long)(size_t)g;
  // D# group0: count=1 | lds_addr | global_addr[56:0] | type=2
  u32x4 g0 = { 1u, lds_addr, (unsigned)ga,
               (unsigned)((ga >> 32) & 0x01ffffffu) | 0x80000000u };
  // D# group1
  unsigned w0 = (1u << 16)            // data_size = 1 (2 bytes)
              | (1u << 20)            // pad_enable
              | (padi << 22)          // pad_interval code
              | (pada << 25);         // pad_amount code
  unsigned d0 = 1u << 30, d1 = 1u << 30;   // huge tensor extents (no OOB clip)
  i32x8 g1 = { (int)w0,
               (int)((d0 & 0xffffu) << 16),                          // dim0 lo16
               (int)(((d0 >> 16) & 0xffffu) | ((d1 & 0xffffu) << 16)),
               (int)(((d1 >> 16) & 0xffffu) | (tile0 << 16)),        // tile_dim0
               (int)(tile1 & 0xffffu),                               // tile_dim1
               (int)stride0,                                         // stride lo32
               0,                                                    // stride hi16
               0 };
  i32x4 z4 = { 0, 0, 0, 0 };
#if __clang_major__ >= 23
  i32x8 z8 = { 0, 0, 0, 0, 0, 0, 0, 0 };
  __builtin_amdgcn_tensor_load_to_lds(g0, g1, z4, z4, z8, 0);
#else
  __builtin_amdgcn_tensor_load_to_lds(g0, g1, z4, z4, 0);
#endif
}
#endif  // USE_TDM

// ---------------------------------------------------------------------------
// Kernel 0: fp32 -> bf16 weight conversion
// ---------------------------------------------------------------------------
__global__ void cvt_bf16_kernel(const float* __restrict__ src,
                                __bf16* __restrict__ dst, int n) {
  for (int i = blockIdx.x * blockDim.x + threadIdx.x; i < n;
       i += gridDim.x * blockDim.x) {
    dst[i] = (__bf16)src[i];
  }
}

// ---------------------------------------------------------------------------
// Kernel 1: GroupNorm over (C/G, T) per (b,g); writes bf16 xn[B][C][T]
// ---------------------------------------------------------------------------
__global__ void groupnorm_kernel(const float* __restrict__ x,
                                 const float* __restrict__ gamma,
                                 const float* __restrict__ beta,
                                 __bf16* __restrict__ xn) {
  constexpr int CPG = Cc / Gg;        // 16 channels per group
  constexpr int NE  = CPG * Tt;       // 16384 elements per group
  int b = blockIdx.x / Gg;
  int g = blockIdx.x % Gg;
  int tid = threadIdx.x;              // 256 threads

  const float* xp = x + (size_t)b * Cc * Tt + (size_t)g * CPG * Tt;
  __bf16* xnp     = xn + (size_t)b * Cc * Tt + (size_t)g * CPG * Tt;

  float s = 0.f, ss = 0.f;
  for (int i = tid; i < NE; i += 256) {
    float v = xp[i];
    s += v;
    ss += v * v;
  }
  __shared__ float rs[256], rq[256];
  rs[tid] = s; rq[tid] = ss;
  __syncthreads();
#pragma unroll
  for (int off = 128; off > 0; off >>= 1) {
    if (tid < off) { rs[tid] += rs[tid + off]; rq[tid] += rq[tid + off]; }
    __syncthreads();
  }
  float mean = rs[0] * (1.0f / NE);
  float var  = rq[0] * (1.0f / NE) - mean * mean;
  float rstd = rsqrtf(var + 1e-5f);

  for (int i = tid; i < NE; i += 256) {
    int c = g * CPG + (i >> 10);      // i / Tt
    float v = (xp[i] - mean) * rstd * gamma[c] + beta[c];
    xnp[i] = (__bf16)v;
  }
}

// ---------------------------------------------------------------------------
// Kernel 2/4: tiled WMMA GEMM  out[b] = A[M,K] * Bm[b][K,N] + bias (+residual)
// Block tile 128x64, K-step 64, 8 waves of 32x32 (2x2 fragments, 2 K-halves).
// A tile DMA'd by the Tensor Data Mover into [m][k] LDS (padded pitch);
// B tile staged transposed [n][k] via in-register 8x8 transposes.
// QSCALE: pre-scale rows < Cc (the Q rows of the QKV output) by 1/sqrt(hd),
// removing the softmax scale from the attention critical path.
// ---------------------------------------------------------------------------
template <int M, int K, bool RES, bool QSCALE>
__global__ void gemm_kernel(const __bf16* __restrict__ A,     // [M][K] bf16
                            const __bf16* __restrict__ Bm,    // [Bn][K][N] bf16
                            const float* __restrict__ bias,   // [M]
                            __bf16* __restrict__ outb,        // [Bn][M][N] bf16 (!RES)
                            float* __restrict__ outf,         // f32 out (RES)
                            const float* __restrict__ resid)  // x (RES)
{
  constexpr int N = 1024;
  constexpr int BM = 128, BN = 64, BK = 64;
  constexpr int LDA = 72, LDB = 72;   // padded LDS strides (16B-aligned rows)

  int nt = blockIdx.x, mt = blockIdx.y, b = blockIdx.z;
  int tid  = threadIdx.x;             // 256
  int lane = tid & 31, wave = tid >> 5;
  int wm = wave >> 1, wn = wave & 1;  // 4x2 wave grid
  int hf = lane >> 4, l15 = lane & 15;

  const __bf16* Bp = Bm + (size_t)b * K * N;
  int m0 = mt * BM, n0 = nt * BN;

  __shared__ alignas(16) __bf16 As[BM * LDA];    // [m][k]
  __shared__ alignas(16) __bf16 Bst[BN * LDB];   // [n][k] (transposed)

  v8f acc[2][2] = {{{}, {}}, {{}, {}}};

  for (int k0 = 0; k0 < K; k0 += BK) {
    __syncthreads();
#if USE_TDM
    // Async DMA of the 128x64 A tile into padded LDS (rows 128B + 16B pad).
    if (wave == 0) {
      tdm_load_2d(&A[(size_t)m0 * K + k0], (unsigned)(size_t)&As[0],
                  /*tile0=*/BK, /*tile1=*/BM, /*stride0=*/K,
                  /*padi=*/4, /*pada=*/3);
    }
#else
    for (int i = tid; i < (BM * BK) / 8; i += 256) {
      int r = i >> 3, cb = (i & 7) * 8;
      *(uint4*)&As[r * LDA + cb] =
          *(const uint4*)&A[(size_t)(m0 + r) * K + k0 + cb];
    }
#endif
    // Stage B tile transposed: 64(k) x 64(n) via 64 8x8 register transposes
    if (tid < 64) {
      int bk = tid & 7, bnb = tid >> 3;
      xpose8x8(&Bp[(size_t)(k0 + bk * 8) * N + n0 + bnb * 8], N,
               &Bst[(bnb * 8) * LDB + bk * 8], LDB);
      if (k0 + BK < K) {  // hint next B tile into cache (global_prefetch_b8)
        __builtin_prefetch(&Bp[(size_t)(k0 + BK + bk * 8) * N + n0 + bnb * 8], 0, 1);
      }
    }
#if USE_TDM
    if (wave == 0) __builtin_amdgcn_s_wait_tensorcnt(0);
#endif
    __syncthreads();

    v16bf a00 = load_frag_a(&As[(wm * 32 +  0) * LDA +  0], LDA, lane);
    v16bf a01 = load_frag_a(&As[(wm * 32 +  0) * LDA + 32], LDA, lane);
    v16bf a10 = load_frag_a(&As[(wm * 32 + 16) * LDA +  0], LDA, lane);
    v16bf a11 = load_frag_a(&As[(wm * 32 + 16) * LDA + 32], LDA, lane);
    v16bf b00 = load_frag_bt(&Bst[(wn * 32 +  0) * LDB +  0], LDB, lane);
    v16bf b01 = load_frag_bt(&Bst[(wn * 32 +  0) * LDB + 32], LDB, lane);
    v16bf b10 = load_frag_bt(&Bst[(wn * 32 + 16) * LDB +  0], LDB, lane);
    v16bf b11 = load_frag_bt(&Bst[(wn * 32 + 16) * LDB + 32], LDB, lane);

    acc[0][0] = wmma_bf16(a00, b00, acc[0][0]);
    acc[0][0] = wmma_bf16(a01, b01, acc[0][0]);
    acc[0][1] = wmma_bf16(a00, b10, acc[0][1]);
    acc[0][1] = wmma_bf16(a01, b11, acc[0][1]);
    acc[1][0] = wmma_bf16(a10, b00, acc[1][0]);
    acc[1][0] = wmma_bf16(a11, b01, acc[1][0]);
    acc[1][1] = wmma_bf16(a10, b10, acc[1][1]);
    acc[1][1] = wmma_bf16(a11, b11, acc[1][1]);
  }

  // Epilogue: C/D layout -> lane holds N=l15, rows r+8*hf
#pragma unroll
  for (int i = 0; i < 2; ++i) {
#pragma unroll
    for (int j = 0; j < 2; ++j) {
      int mb = m0 + wm * 32 + i * 16 + hf * 8;
      int n  = n0 + wn * 32 + j * 16 + l15;
#pragma unroll
      for (int r = 0; r < 8; ++r) {
        int row = mb + r;
        float v = acc[i][j][r] + bias[row];
        if constexpr (QSCALE) {
          if (row < Cc) v *= 0.125f;  // fold 1/sqrt(hd) into Q rows
        }
        size_t idx = (size_t)b * M * N + (size_t)row * N + n;
        if constexpr (RES) {
          outf[idx] = resid[idx] + v;
        } else {
          outb[idx] = (__bf16)v;
        }
      }
    }
  }
}

// ---------------------------------------------------------------------------
// Kernel 3: flash attention per (b,h). Q,K,V are rows of qkv[b][3C][T] (bf16).
// Q is pre-scaled by 1/sqrt(hd) in the QKV epilogue, so S = Q'K^T directly.
// Block: 128 threads = 4 waves; 64-query tile, 64-key chunks, online softmax.
// Q,K staged transposed ([t][d], [s][d]); V tile DMA'd by TDM into [d][s].
// ---------------------------------------------------------------------------
__global__ void attn_kernel(const __bf16* __restrict__ qkv,
                            __bf16* __restrict__ aout) {
  constexpr int LDT = 72;   // padded stride
  int bh = blockIdx.x;                // 0..63
  int b = bh >> 3, h = bh & 7;
  int t0 = blockIdx.y * 64;           // query tile base
  int tid  = threadIdx.x;             // 128
  int lane = tid & 31, wave = tid >> 5;
  int hf = lane >> 4, l15 = lane & 15;

  __shared__ alignas(16) __bf16 Qst[64 * LDT];       // [t][d]
  __shared__ alignas(16) __bf16 Kst[64 * LDT];       // [s][d]
  __shared__ alignas(16) __bf16 Vs[64 * LDT];        // [d][s]
  __shared__ alignas(16) __bf16 Ps[4 * 16 * LDT];    // per-wave P (A layout [t][s])

  const size_t bb = (size_t)b * (3 * Cc) * Tt;
  const __bf16* qp = qkv + bb + (size_t)(0 * Cc + h * HDd) * Tt;
  const __bf16* kp = qkv + bb + (size_t)(1 * Cc + h * HDd) * Tt;
  const __bf16* vp = qkv + bb + (size_t)(2 * Cc + h * HDd) * Tt;

  // Stage Q transposed: global [d][t] -> LDS [t][d], 64 8x8 blocks
  if (tid < 64) {
    int bd = tid & 7, bt = tid >> 3;
    xpose8x8(&qp[(size_t)(bd * 8) * Tt + t0 + bt * 8], Tt,
             &Qst[(bt * 8) * LDT + bd * 8], LDT);
  }
  __syncthreads();

  float mrow[8], lrow[8];
  v8f of[4] = {{}, {}, {}, {}};
#pragma unroll
  for (int r = 0; r < 8; ++r) { mrow[r] = -3.0e38f; lrow[r] = 0.f; }

  for (int s0 = 0; s0 < Tt; s0 += 64) {
    __syncthreads();
    // Stage K transposed: [d][s] -> [s][d]
    if (tid < 64) {
      int bd = tid & 7, bs = tid >> 3;
      xpose8x8(&kp[(size_t)(bd * 8) * Tt + s0 + bs * 8], Tt,
               &Kst[(bs * 8) * LDT + bd * 8], LDT);
    }
#if USE_TDM
    // Async DMA of the 64x64 V tile into padded LDS (rows 128B + 16B pad).
    if (wave == 0) {
      tdm_load_2d(&vp[s0], (unsigned)(size_t)&Vs[0],
                  /*tile0=*/64, /*tile1=*/64, /*stride0=*/Tt,
                  /*padi=*/4, /*pada=*/3);
      __builtin_amdgcn_s_wait_tensorcnt(0);
    }
#else
    for (int i = tid; i < 512; i += 128) {
      int r = i >> 3, cb = (i & 7) * 8;
      *(uint4*)&Vs[r * LDT + cb] =
          *(const uint4*)&vp[(size_t)r * Tt + s0 + cb];
    }
#endif
    __syncthreads();

    // Q A-fragments re-loaded from resident LDS each chunk (avoids spills)
    v16bf qa0 = load_frag_a(&Qst[(wave * 16) * LDT +  0], LDT, lane);
    v16bf qa1 = load_frag_a(&Qst[(wave * 16) * LDT + 32], LDT, lane);

    // S = Q' K^T : M=16 q rows, N=64 s cols, K=d=64 (scale pre-folded into Q)
    v8f sf[4];
#pragma unroll
    for (int nf = 0; nf < 4; ++nf) {
      v8f c = {};
      c = wmma_bf16(qa0, load_frag_bt(&Kst[(nf * 16) * LDT +  0], LDT, lane), c);
      c = wmma_bf16(qa1, load_frag_bt(&Kst[(nf * 16) * LDT + 32], LDT, lane), c);
      sf[nf] = c;
    }

    // Online softmax over this 64-column chunk (row-wise over N)
#pragma unroll
    for (int r = 0; r < 8; ++r) {
      float mx = fmaxf(fmaxf(sf[0][r], sf[1][r]), fmaxf(sf[2][r], sf[3][r]));
#pragma unroll
      for (int off = 1; off < 16; off <<= 1)
        mx = fmaxf(mx, __shfl_xor(mx, off, 32));
      float mnew  = fmaxf(mrow[r], mx);
      float alpha = __expf(mrow[r] - mnew);
      float rsum  = 0.f;
#pragma unroll
      for (int nf = 0; nf < 4; ++nf) {
        float p = __expf(sf[nf][r] - mnew);
        sf[nf][r] = p;
        rsum += p;
      }
#pragma unroll
      for (int off = 1; off < 16; off <<= 1)
        rsum += __shfl_xor(rsum, off, 32);
      lrow[r] = lrow[r] * alpha + rsum;
      mrow[r] = mnew;
#pragma unroll
      for (int nf = 0; nf < 4; ++nf) of[nf][r] *= alpha;
    }

    // P (C-frag layout) -> per-wave LDS (A layout [t][s]) -> A-fragments
    __bf16* pw = &Ps[wave * 16 * LDT];
#pragma unroll
    for (int nf = 0; nf < 4; ++nf) {
#pragma unroll
      for (int r = 0; r < 8; ++r) {
        pw[(r + 8 * hf) * LDT + nf * 16 + l15] = (__bf16)sf[nf][r];
      }
    }
    v16bf pa0 = load_frag_a(&pw[0], LDT, lane);
    v16bf pa1 = load_frag_a(&pw[32], LDT, lane);

    // O += P * V^T : M=t, N=d, K=s ; B[k=s][n=d] = Vs[d][s] (K-contiguous gather)
#pragma unroll
    for (int nf = 0; nf < 4; ++nf) {
      of[nf] = wmma_bf16(pa0, load_frag_bt(&Vs[(nf * 16) * LDT +  0], LDT, lane), of[nf]);
      of[nf] = wmma_bf16(pa1, load_frag_bt(&Vs[(nf * 16) * LDT + 32], LDT, lane), of[nf]);
    }
  }

  // Normalize and store A out: aout[b][h*64 + d][t] (bf16)
#pragma unroll
  for (int nf = 0; nf < 4; ++nf) {
#pragma unroll
    for (int r = 0; r < 8; ++r) {
      int t = t0 + wave * 16 + r + 8 * hf;
      int d = nf * 16 + l15;
      float v = of[nf][r] / lrow[r];
      aout[(size_t)b * Cc * Tt + (size_t)(h * HDd + d) * Tt + t] = (__bf16)v;
    }
  }
}

// ---------------------------------------------------------------------------
// Launch
// ---------------------------------------------------------------------------
extern "C" void kernel_launch(void* const* d_in, const int* in_sizes, int n_in,
                              void* d_out, int out_size, void* d_ws, size_t ws_size,
                              hipStream_t stream) {
  (void)in_sizes; (void)n_in; (void)out_size; (void)ws_size;
  const float* x        = (const float*)d_in[0];
  const float* gn_gamma = (const float*)d_in[1];
  const float* gn_beta  = (const float*)d_in[2];
  const float* w_qkv    = (const float*)d_in[3];
  const float* b_qkv    = (const float*)d_in[4];
  const float* w_proj   = (const float*)d_in[5];
  const float* b_proj   = (const float*)d_in[6];
  float* out = (float*)d_out;

  // Workspace layout (bf16 buffers)
  char* ws = (char*)d_ws;
  size_t off = 0;
  __bf16* xn_b   = (__bf16*)(ws + off); off += (size_t)Bn * Cc * Tt * 2;        // 8 MB
  __bf16* wqkv_b = (__bf16*)(ws + off); off += (size_t)3 * Cc * Cc * 2;         // 1.5 MB
  __bf16* wprj_b = (__bf16*)(ws + off); off += (size_t)Cc * Cc * 2;             // 0.5 MB
  __bf16* qkv_b  = (__bf16*)(ws + off); off += (size_t)Bn * 3 * Cc * Tt * 2;    // 24 MB
  __bf16* a_b    = (__bf16*)(ws + off); off += (size_t)Bn * Cc * Tt * 2;        // 8 MB

  // 0) weight conversion
  cvt_bf16_kernel<<<1024, 256, 0, stream>>>(w_qkv, wqkv_b, 3 * Cc * Cc);
  cvt_bf16_kernel<<<512, 256, 0, stream>>>(w_proj, wprj_b, Cc * Cc);

  // 1) GroupNorm -> bf16 activations
  groupnorm_kernel<<<Bn * Gg, 256, 0, stream>>>(x, gn_gamma, gn_beta, xn_b);

  // 2) QKV = w_qkv * xn + b_qkv (Q rows pre-scaled by 1/sqrt(hd))
  gemm_kernel<3 * Cc, Cc, false, true>
      <<<dim3(Tt / 64, (3 * Cc) / 128, Bn), 256, 0, stream>>>(
          wqkv_b, xn_b, b_qkv, qkv_b, nullptr, nullptr);

  // 3) Flash attention per (b, h), 64-query tiles
  attn_kernel<<<dim3(Bn * NHh, Tt / 64), 128, 0, stream>>>(qkv_b, a_b);

  // 4) out = x + w_proj * a + b_proj   (M=512, N=1024, K=512 per batch)
  gemm_kernel<Cc, Cc, true, false>
      <<<dim3(Tt / 64, Cc / 128, Bn), 256, 0, stream>>>(
          wprj_b, a_b, b_proj, nullptr, out, x);
}
